// Router_80556406603830
// MI455X (gfx1250) — matrix operand order repaired and checked by reading, not compiled, not run on hardware
//
#include <hip/hip_runtime.h>
#include <hip/hip_bf16.h>
#include <math.h>

#define EMB     4096
#define NEXP    64
#define NTOK    16384
#define WAVES   8                      // waves per block
#define ROWS_PER_WAVE 16
#define KSPLIT  2                      // K halves, one per wave group
#define SUBTILES (WAVES / KSPLIT)      // 4 token subtiles per block
#define ROWS_PER_BLOCK (SUBTILES * ROWS_PER_WAVE)   // 64 tokens per block
#define KSTEP   32
#define NKSTEPS (EMB / KSTEP)          // 128 total
#define KSTEPS_PER_WAVE (NKSTEPS / KSPLIT)          // 64 per wave
#define NTILES  (NEXP / 16)            // 4 expert tiles of 16

typedef __attribute__((ext_vector_type(16))) __bf16 v16bf;
typedef __attribute__((ext_vector_type(8)))  float  v8f;

// ---------------------------------------------------------------------------
// W prepack: f32 [64,4096] -> bf16 fragments in exact WMMA SRC1 (B) layout.
// Fragment addressing: elem = ((kb*NTILES + t)*32 + lane)*16 + ki
//   kb = k-step (k/32), t = expert tile, lane = sel*16 + (expert%16),
//   sel = (k%16)/8, ki = (k%8) + 8*((k%32)/16). Lane n holds 16 contiguous bf16 = 32B.
// ---------------------------------------------------------------------------

__device__ __forceinline__ unsigned f2bf_rne(float f) {
    unsigned u = __float_as_uint(f);
    u += 0x7FFFu + ((u >> 16) & 1u);
    return u >> 16;
}

__global__ __launch_bounds__(256) void prepack_w_kernel(
    const float* __restrict__ W,          // [NEXP, EMB]
    unsigned* __restrict__ wb_u32)        // bf16 fragments, viewed as u32 pairs
{
    const int tid = blockIdx.x * blockDim.x + threadIdx.x;
    const int e     = tid >> 11;          // 2048 k-pairs per expert
    const int kpair = tid & 2047;
    const int k     = kpair * 2;

    const int t    = e >> 4;
    const int n15  = e & 15;
    const int kb   = k >> 5;
    const int sel  = (k >> 3) & 1;
    const int ki   = (k & 7) | (((k >> 4) & 1) << 3);  // even
    const int lane = sel * 16 + n15;

    const float a = W[(size_t)e * EMB + k];
    const float b = W[(size_t)e * EMB + k + 1];
    const unsigned packed = f2bf_rne(a) | (f2bf_rne(b) << 16);

    const int elem = ((kb * NTILES + t) * 32 + lane) * 16 + ki;  // even
    wb_u32[elem >> 1] = packed;
}

// ---------------------------------------------------------------------------
// Main router kernel: bf16 WMMA GEMM (K split across wave pairs, software-
// pipelined loads) + top-2 softmax
// ---------------------------------------------------------------------------

// Pack two f32 into a bf16 pair (round-half-away) with one v_perm_b32.
__device__ __forceinline__ unsigned pack_bf16_pair(float lo, float hi) {
    const unsigned ua = __float_as_uint(lo) + 0x8000u;
    const unsigned ub = __float_as_uint(hi) + 0x8000u;
    return __builtin_amdgcn_perm(ub, ua, 0x07060302u);  // {bf(hi), bf(lo)}
}

union BF16x16u {
    v16bf v;
    unsigned d[8];
};

// Raw A data for one k-step: x[k..k+7] and x[k+16..k+23] (k includes sel*8).
struct ARaw {
    float4 f0, f1, f2, f3;
};

__device__ __forceinline__ ARaw loadARaw(const float* __restrict__ p) {
    ARaw r;
    r.f0 = ((const float4*)p)[0];
    r.f1 = ((const float4*)p)[1];
    r.f2 = ((const float4*)(p + 16))[0];
    r.f3 = ((const float4*)(p + 16))[1];
    return r;
}

__device__ __forceinline__ v16bf packA(const ARaw& a) {
    BF16x16u r;
    r.d[0] = pack_bf16_pair(a.f0.x, a.f0.y);
    r.d[1] = pack_bf16_pair(a.f0.z, a.f0.w);
    r.d[2] = pack_bf16_pair(a.f1.x, a.f1.y);
    r.d[3] = pack_bf16_pair(a.f1.z, a.f1.w);
    r.d[4] = pack_bf16_pair(a.f2.x, a.f2.y);
    r.d[5] = pack_bf16_pair(a.f2.z, a.f2.w);
    r.d[6] = pack_bf16_pair(a.f3.x, a.f3.y);
    r.d[7] = pack_bf16_pair(a.f3.z, a.f3.w);
    return r.v;
}

// B fragment: 32 contiguous bytes of prepacked bf16.
__device__ __forceinline__ v16bf loadB(const unsigned short* __restrict__ p) {
    union { v16bf v; uint4 q[2]; } r;
    r.q[0] = ((const uint4*)p)[0];
    r.q[1] = ((const uint4*)p)[1];
    return r.v;
}

__global__ __launch_bounds__(256) void router_topk_kernel(
    const float* __restrict__ x,              // [NTOK, EMB] f32
    const unsigned short* __restrict__ wb,    // prepacked bf16 W fragments
    const float* __restrict__ bias,           // [NEXP]
    int*   __restrict__ idx_out,              // [NTOK, 2]
    float* __restrict__ prob_out)             // [NTOK, 2]
{
    // [k-half][token subtile][token row][expert], padded to 68 (stride 4 mod 64 banks)
    __shared__ float logits_lds[KSPLIT][SUBTILES][ROWS_PER_WAVE][NEXP + 4];

    const int lane   = threadIdx.x & 31;
    const int wave   = threadIdx.x >> 5;
    const int lane15 = lane & 15;
    const int sel    = lane >> 4;
    const int grp    = wave >> 2;             // which K half this wave computes
    const int sub    = wave & (SUBTILES - 1); // which 16-token subtile
    const int rowBase = blockIdx.x * ROWS_PER_BLOCK + sub * ROWS_PER_WAVE;
    const int kbElem  = grp * (EMB / KSPLIT); // element offset of this K half

    const float* __restrict__ xrow =
        x + (size_t)(rowBase + lane15) * EMB + kbElem + sel * 8;
    // B fragments for this K half; advance 2048 elements (4 tiles * 512) per k-step
    const unsigned short* __restrict__ bptr =
        wb + (size_t)lane * 16 + (size_t)grp * KSTEPS_PER_WAVE * (NTILES * 512);

    v8f acc0 = {}, acc1 = {}, acc2 = {}, acc3 = {};

    // ---- software pipeline: stage 0 preload ----
    ARaw  ar = loadARaw(xrow);
    v16bf b0 = loadB(bptr + 0 * 512);
    v16bf b1 = loadB(bptr + 1 * 512);
    v16bf b2 = loadB(bptr + 2 * 512);
    v16bf b3 = loadB(bptr + 3 * 512);

    for (int kb = 0; kb < KSTEPS_PER_WAVE; ++kb) {
        // Issue next iteration's loads first (clamped on the last step so the
        // addresses stay in-bounds with uniform control flow / full EXEC).
        const int nk = (kb + 1 < KSTEPS_PER_WAVE) ? (kb + 1) : kb;
        const ARaw nar = loadARaw(xrow + nk * KSTEP);
        const unsigned short* nbp = bptr + (size_t)nk * (NTILES * 512);
        const v16bf nb0 = loadB(nbp + 0 * 512);
        const v16bf nb1 = loadB(nbp + 1 * 512);
        const v16bf nb2 = loadB(nbp + 2 * 512);
        const v16bf nb3 = loadB(nbp + 3 * 512);

        // Consume current operands while next loads are in flight.
        const v16bf a = packA(ar);
        acc0 = __builtin_amdgcn_wmma_f32_16x16x32_bf16(false, a, false, b0, (short)0, acc0, false, false);
        acc1 = __builtin_amdgcn_wmma_f32_16x16x32_bf16(false, a, false, b1, (short)0, acc1, false, false);
        acc2 = __builtin_amdgcn_wmma_f32_16x16x32_bf16(false, a, false, b2, (short)0, acc2, false, false);
        acc3 = __builtin_amdgcn_wmma_f32_16x16x32_bf16(false, a, false, b3, (short)0, acc3, false, false);

        ar = nar;
        b0 = nb0; b1 = nb1; b2 = nb2; b3 = nb3;
    }

    // Bias added only by the low-K wave group (avoid double count).
    const float bscale = (grp == 0) ? 1.0f : 0.0f;
    const float bv0 = bscale * bias[ 0 + lane15];
    const float bv1 = bscale * bias[16 + lane15];
    const float bv2 = bscale * bias[32 + lane15];
    const float bv3 = bscale * bias[48 + lane15];

    // C/D layout: VGPR i of lanes 0-15 is token row i; lanes 16-31 row i+8.
    #pragma unroll
    for (int i = 0; i < 8; ++i) {
        const int m = sel ? (i + 8) : i;
        logits_lds[grp][sub][m][ 0 + lane15] = acc0[i] + bv0;
        logits_lds[grp][sub][m][16 + lane15] = acc1[i] + bv1;
        logits_lds[grp][sub][m][32 + lane15] = acc2[i] + bv2;
        logits_lds[grp][sub][m][48 + lane15] = acc3[i] + bv3;
    }

    __syncthreads();

    // Top-2 + softmax: low-K waves, lanes 0-15, one token each. Strict > keeps
    // the first index on ties, matching jax.lax.top_k.
    if (grp == 0 && lane < 16) {
        const float* rowA = &logits_lds[0][sub][lane][0];
        const float* rowB = &logits_lds[1][sub][lane][0];
        float v0 = -INFINITY, v1 = -INFINITY;
        int   i0 = 0,         i1 = 0;
        #pragma unroll 8
        for (int e = 0; e < NEXP; ++e) {
            const float f = rowA[e] + rowB[e];
            if (f > v0)      { v1 = v0; i1 = i0; v0 = f; i0 = e; }
            else if (f > v1) { v1 = f;  i1 = e; }
        }
        const float e1 = __expf(v1 - v0);   // v1 <= v0: stable
        const float p0 = 1.0f / (1.0f + e1);
        const float p1 = e1 * p0;

        const int tok = rowBase + lane;
        idx_out[tok * 2 + 0]  = i0;
        idx_out[tok * 2 + 1]  = i1;
        prob_out[tok * 2 + 0] = p0;
        prob_out[tok * 2 + 1] = p1;
    }
}

extern "C" void kernel_launch(void* const* d_in, const int* in_sizes, int n_in,
                              void* d_out, int out_size, void* d_ws, size_t ws_size,
                              hipStream_t stream) {
    const float* x = (const float*)d_in[0];   // [16384, 4096]
    const float* W = (const float*)d_in[1];   // [64, 4096]
    const float* b = (const float*)d_in[2];   // [64]

    // d_ws: 512 KB of prepacked bf16 W fragments.
    unsigned*       wb_u32 = (unsigned*)d_ws;
    unsigned short* wb     = (unsigned short*)d_ws;

    int*   idx_out  = (int*)d_out;
    float* prob_out = (float*)d_out + (size_t)NTOK * 2;

    // 1) Prepack W -> bf16 fragments (64*4096/2 pairs / 256 threads = 512 blocks).
    hipLaunchKernelGGL(prepack_w_kernel, dim3((NEXP * EMB / 2) / 256), dim3(256),
                       0, stream, W, wb_u32);

    // 2) Router GEMM (K split, pipelined) + top-2 + softmax.
    hipLaunchKernelGGL(router_topk_kernel, dim3(NTOK / ROWS_PER_BLOCK), dim3(WAVES * 32),
                       0, stream, x, wb, b, idx_out, prob_out);
}
